// minLSTM_5111011082249
// MI455X (gfx1250) — matrix operand order repaired and checked by reading, not compiled, not run on hardware
//
#include <hip/hip_runtime.h>
#include <math.h>

// ---------------- problem constants ----------------
#define B_   4
#define T_   4096
#define D_   1024
#define M_   (B_ * T_)          // 16384 rows of X
#define KDIM D_
#define NDIM D_

// GEMM tiling
#define BM 128
#define BN 128
#define KS 32
#define NK (KDIM / KS)          // 32 K-steps

// scan chunking
#define CS 128                  // timesteps per chunk
#define NC (T_ / CS)            // 32 chunks per sequence

typedef __bf16 bf16_t;
typedef __attribute__((ext_vector_type(16))) __bf16 v16bf;
typedef __attribute__((ext_vector_type(8)))  __bf16 v8bf;
typedef __attribute__((ext_vector_type(8)))  float  v8f;
typedef __attribute__((ext_vector_type(4)))  int    v4i;

// CDNA5 async global->LDS path (guarded: falls back to sync copy if absent)
#if defined(__gfx1250__) && __has_builtin(__builtin_amdgcn_global_load_async_to_lds_b128)
#define USE_ASYNC_LDS 1
#else
#define USE_ASYNC_LDS 0
#endif

// builtin signature (from hipcc diagnostic): (v4i AS1*, v4i AS3*, Imm, Imm)
typedef __attribute__((address_space(1))) v4i gv4i_t;
typedef __attribute__((address_space(3))) v4i lv4i_t;

static __device__ __forceinline__ void wait_async0() {
#if defined(__gfx1250__) && __has_builtin(__builtin_amdgcn_s_wait_asynccnt)
    __builtin_amdgcn_s_wait_asynccnt(0);
#elif defined(__gfx1250__)
    asm volatile("s_wait_asynccnt 0x0" ::: "memory");
#endif
}

// ---------------- bf16 split helpers ----------------
static __device__ __forceinline__ bf16_t f2bf_rne(float f) {
    unsigned u = __builtin_bit_cast(unsigned, f);
    unsigned r = u + 0x7FFFu + ((u >> 16) & 1u);   // round to nearest even
    unsigned short hs = (unsigned short)(r >> 16);
    return __builtin_bit_cast(bf16_t, hs);
}
static __device__ __forceinline__ float bf2f(bf16_t h) {
    unsigned short hs = __builtin_bit_cast(unsigned short, h);
    unsigned u = ((unsigned)hs) << 16;
    return __builtin_bit_cast(float, u);
}
static __device__ __forceinline__ void split_bf16(float f, bf16_t& hi, bf16_t& lo) {
    hi = f2bf_rne(f);
    lo = f2bf_rne(f - bf2f(hi));
}

// =====================================================================
// Pre-pass A: X fp32 -> bf16 hi/lo planes, row-major [M_][KDIM]
// =====================================================================
__global__ __launch_bounds__(256) void minlstm_convert_x(
    const float* __restrict__ X, bf16_t* __restrict__ Xhi, bf16_t* __restrict__ Xlo)
{
    size_t base = ((size_t)blockIdx.x * 256 + threadIdx.x) * 8;
    float v[8];
    *(float4*)(v + 0) = *(const float4*)(X + base + 0);
    *(float4*)(v + 4) = *(const float4*)(X + base + 4);
    bf16_t h[8], l[8];
#pragma unroll
    for (int q = 0; q < 8; ++q) split_bf16(v[q], h[q], l[q]);
    *(v8bf*)(Xhi + base) = *(v8bf*)h;
    *(v8bf*)(Xlo + base) = *(v8bf*)l;
}

// =====================================================================
// Pre-pass B: W fp32 [K][N] -> transposed bf16 hi/lo [gate][N][K]
//   (fragment-friendly: B fragments become contiguous 32B runs in K)
// =====================================================================
__global__ __launch_bounds__(256) void minlstm_convert_w(
    const float* __restrict__ Wf, const float* __restrict__ Wi,
    const float* __restrict__ Wh,
    bf16_t* __restrict__ WThi, bf16_t* __restrict__ WTlo)
{
    const int gate = blockIdx.z;
    const float* __restrict__ W = (gate == 0) ? Wf : ((gate == 1) ? Wi : Wh);
    bf16_t* __restrict__ thi = WThi + (size_t)gate * NDIM * KDIM;
    bf16_t* __restrict__ tlo = WTlo + (size_t)gate * NDIM * KDIM;

    __shared__ float tile[32][33];
    const int k0 = blockIdx.x * 32, n0 = blockIdx.y * 32;
    const int tx = threadIdx.x & 31, ty = threadIdx.x >> 5;   // ty 0..7
#pragma unroll
    for (int i = 0; i < 4; ++i)
        tile[ty + 8 * i][tx] = W[(size_t)(k0 + ty + 8 * i) * NDIM + n0 + tx];
    __syncthreads();
#pragma unroll
    for (int i = 0; i < 4; ++i) {
        int nl = ty + 8 * i;                       // local n
        float f = tile[tx][nl];                    // = W[k0+tx][n0+nl]
        bf16_t h, l; split_bf16(f, h, l);
        size_t o = (size_t)(n0 + nl) * KDIM + k0 + tx;   // coalesced in k
        thi[o] = h; tlo[o] = l;
    }
}

// =====================================================================
// Kernel 1: 3-gate GEMM  k = X @ W + b  via bf16-split WMMA
//   double-buffered LDS fed by GLOBAL_LOAD_ASYNC_TO_LDS (when available)
//   grid = (NDIM/BN, M_/BM, 3 gates), block = 256 threads (8 wave32)
// =====================================================================
// copy 128 rows x 32 bf16 (64B/row) from global (row stride KDIM) into LDS
static __device__ __forceinline__ void stage_plane(
    const bf16_t* __restrict__ g, bf16_t* l, int tid)
{
#pragma unroll
    for (int i = 0; i < 2; ++i) {
        int id  = i * 256 + tid;        // 512 x 16B lane-ops
        int row = id >> 2;
        int seg = (id & 3) * 8;
        const bf16_t* src = g + (size_t)row * KDIM + seg;
        bf16_t* dst = l + row * KS + seg;
#if USE_ASYNC_LDS
        __builtin_amdgcn_global_load_async_to_lds_b128(
            (gv4i_t*)(void*)(bf16_t*)src, (lv4i_t*)(void*)dst, 0, 0);
#else
        *(v8bf*)dst = *(const v8bf*)src;
#endif
    }
}

__global__ __launch_bounds__(256) void minlstm_gate_gemm(
    const bf16_t* __restrict__ Xhi, const bf16_t* __restrict__ Xlo,
    const bf16_t* __restrict__ WThi, const bf16_t* __restrict__ WTlo,
    const float* __restrict__ bf, const float* __restrict__ bi,
    const float* __restrict__ bh,
    float* __restrict__ Kout /* [3][M_][NDIM] */)
{
    const int gate = blockIdx.z;
    const bf16_t* __restrict__ Bhig = WThi + (size_t)gate * NDIM * KDIM;
    const bf16_t* __restrict__ Blog = WTlo + (size_t)gate * NDIM * KDIM;
    const float* __restrict__ bm = (gate == 0) ? bf : ((gate == 1) ? bi : bh);
    float* __restrict__ out = Kout + (size_t)gate * M_ * NDIM;

    const int n0   = blockIdx.x * BN;
    const int m0   = blockIdx.y * BM;
    const int tid  = threadIdx.x;
    const int lane = tid & 31;
    const int wv   = tid >> 5;         // wave 0..7
    const int wm0  = (wv >> 2) * 64;   // 2 wave rows
    const int wn0  = (wv & 3) * 32;    // 4 wave cols

    __shared__ alignas(16) bf16_t sA[2][2][BM * KS];  // [buf][hi/lo]
    __shared__ alignas(16) bf16_t sB[2][2][BN * KS];  // [buf][hi/lo], [n][k]

    const bf16_t* __restrict__ gXh = Xhi + (size_t)m0 * KDIM;
    const bf16_t* __restrict__ gXl = Xlo + (size_t)m0 * KDIM;
    const bf16_t* __restrict__ gBh = Bhig + (size_t)n0 * KDIM;
    const bf16_t* __restrict__ gBl = Blog + (size_t)n0 * KDIM;

    v8f acc[4][2];
#pragma unroll
    for (int i = 0; i < 4; ++i)
#pragma unroll
        for (int j = 0; j < 2; ++j)
#pragma unroll
            for (int q = 0; q < 8; ++q) acc[i][j][q] = 0.0f;

    const int arow = lane & 15;
    const int akb  = (lane < 16) ? 0 : 16;   // A K-half base (contiguous, matches B)
    const int bcol = lane & 15;
    const int bkb  = (lane < 16) ? 0 : 16;   // B K-half base (contiguous, matches A)

    // prologue: stage tile 0 into buffer 0
    stage_plane(gXh, sA[0][0], tid);
    stage_plane(gXl, sA[0][1], tid);
    stage_plane(gBh, sB[0][0], tid);
    stage_plane(gBl, sB[0][1], tid);

    for (int ks = 0; ks < NK; ++ks) {
        const int buf = ks & 1;
        wait_async0();            // my async copies into buf are done
        __syncthreads();          // everyone's copies visible; buf^1 free

        if (ks + 1 < NK) {        // prefetch next tile into other buffer
            const int kn = (ks + 1) * KS;
            stage_plane(gXh + kn, sA[buf ^ 1][0], tid);
            stage_plane(gXl + kn, sA[buf ^ 1][1], tid);
            stage_plane(gBh + kn, sB[buf ^ 1][0], tid);
            stage_plane(gBl + kn, sB[buf ^ 1][1], tid);
        }

        // ---- build fragments from LDS ----
        v16bf ah[4], al[4], bhf[2], blf[2];
#pragma unroll
        for (int i = 0; i < 4; ++i) {
            int row = wm0 + 16 * i + arow;
            v8bf h0 = *(const v8bf*)(sA[buf][0] + row * KS + akb);
            v8bf h1 = *(const v8bf*)(sA[buf][0] + row * KS + akb + 8);
            v8bf l0 = *(const v8bf*)(sA[buf][1] + row * KS + akb);
            v8bf l1 = *(const v8bf*)(sA[buf][1] + row * KS + akb + 8);
#pragma unroll
            for (int q = 0; q < 8; ++q) {
                ah[i][q] = h0[q]; ah[i][8 + q] = h1[q];
                al[i][q] = l0[q]; al[i][8 + q] = l1[q];
            }
        }
#pragma unroll
        for (int j = 0; j < 2; ++j) {
            int col = wn0 + 16 * j + bcol;
            bhf[j] = *(const v16bf*)(sB[buf][0] + col * KS + bkb);
            blf[j] = *(const v16bf*)(sB[buf][1] + col * KS + bkb);
        }

        // ---- 4x2 tiles x 3 split products = 24 WMMAs per K-step ----
#pragma unroll
        for (int i = 0; i < 4; ++i) {
#pragma unroll
            for (int j = 0; j < 2; ++j) {
                acc[i][j] = __builtin_amdgcn_wmma_f32_16x16x32_bf16(
                    false, ah[i], false, bhf[j], (short)0, acc[i][j], false, false);
                acc[i][j] = __builtin_amdgcn_wmma_f32_16x16x32_bf16(
                    false, al[i], false, bhf[j], (short)0, acc[i][j], false, false);
                acc[i][j] = __builtin_amdgcn_wmma_f32_16x16x32_bf16(
                    false, ah[i], false, blf[j], (short)0, acc[i][j], false, false);
            }
        }
    }

    // ---- epilogue: bias add + store (ISA C/D layout: VGPR r -> M=r(+8)) ----
    const int rbase = (lane < 16) ? 0 : 8;
#pragma unroll
    for (int i = 0; i < 4; ++i) {
#pragma unroll
        for (int j = 0; j < 2; ++j) {
            int ng = n0 + wn0 + 16 * j + bcol;
            float bias = bm[ng];
#pragma unroll
            for (int r = 0; r < 8; ++r) {
                int mg = m0 + wm0 + 16 * i + r + rbase;
                out[(size_t)mg * NDIM + ng] = acc[i][j][r] + bias;
            }
        }
    }
}

// =====================================================================
// Scan math helpers (log-space minLSTM recurrence)
//   x_t = LSE(a_t + x_{t-1}, b_t);  a = log f, b = log i + log g(k_h)
// =====================================================================
static __device__ __forceinline__ float sp(float x) {           // softplus, stable
    return fmaxf(x, 0.0f) + log1pf(expf(-fabsf(x)));
}
static __device__ __forceinline__ float lse2(float u, float v) { // logaddexp, -inf safe
    float m  = fmaxf(u, v);
    float mn = fminf(u, v);
    float r  = m + log1pf(expf(mn - m));
    return (mn == -INFINITY) ? m : r;
}
static __device__ __forceinline__ void gate_logs(float kf, float ki, float kh,
                                                 float& a, float& b) {
    float diff  = sp(-kf) - sp(-ki);
    a           = -sp(diff);                 // log f'
    float log_i = -sp(-diff);                // log i'
    float log_g = (kh >= 0.0f) ? logf(kh + 0.5f) : -sp(-kh);
    b = log_i + log_g;
}

// ---- pass 1: per-chunk composition (A,B): x_end = LSE(A + x_start, B) ----
__global__ __launch_bounds__(256) void minlstm_scan_chunk(
    const float* __restrict__ kf, const float* __restrict__ ki,
    const float* __restrict__ kh,
    float* __restrict__ Asum, float* __restrict__ Bsum)
{
    const int d = blockIdx.x * 256 + threadIdx.x;
    const int c = blockIdx.y;
    const int b = blockIdx.z;
    float A = 0.0f, Bv = -INFINITY;
    size_t base = ((size_t)b * T_ + (size_t)c * CS) * D_ + d;
    for (int t = 0; t < CS; ++t) {
        size_t idx = base + (size_t)t * D_;
        float a, vb;
        gate_logs(kf[idx], ki[idx], kh[idx], a, vb);
        Bv = lse2(a + Bv, vb);
        A += a;
    }
    size_t o = ((size_t)b * NC + c) * D_ + d;
    Asum[o] = A; Bsum[o] = Bv;
}

// ---- pass 2: scan the NC chunk summaries per channel -> prefix states ----
__global__ __launch_bounds__(256) void minlstm_scan_combine(
    const float* __restrict__ h0,
    const float* __restrict__ Asum, const float* __restrict__ Bsum,
    float* __restrict__ prefix)
{
    int id = blockIdx.x * 256 + threadIdx.x;
    if (id >= B_ * D_) return;
    int b = id / D_, d = id % D_;
    float x = logf(h0[(size_t)b * D_ + d]);
    for (int c = 0; c < NC; ++c) {
        size_t o = ((size_t)b * NC + c) * D_ + d;
        prefix[o] = x;
        x = lse2(Asum[o] + x, Bsum[o]);
    }
}

// ---- pass 3: replay each chunk from its prefix, emit h_t = exp(x_t) ----
__global__ __launch_bounds__(256) void minlstm_scan_apply(
    const float* __restrict__ kf, const float* __restrict__ ki,
    const float* __restrict__ kh,
    const float* __restrict__ prefix, float* __restrict__ out)
{
    const int d = blockIdx.x * 256 + threadIdx.x;
    const int c = blockIdx.y;
    const int b = blockIdx.z;
    float x = prefix[((size_t)b * NC + c) * D_ + d];
    size_t base = ((size_t)b * T_ + (size_t)c * CS) * D_ + d;
    for (int t = 0; t < CS; ++t) {
        size_t idx = base + (size_t)t * D_;
        float a, vb;
        gate_logs(kf[idx], ki[idx], kh[idx], a, vb);
        x = lse2(a + x, vb);
        out[idx] = expf(x);
    }
}

// =====================================================================
// launcher
// =====================================================================
extern "C" void kernel_launch(void* const* d_in, const int* in_sizes, int n_in,
                              void* d_out, int out_size, void* d_ws, size_t ws_size,
                              hipStream_t stream) {
    const float* x  = (const float*)d_in[0];
    const float* h0 = (const float*)d_in[1];
    const float* Wf = (const float*)d_in[2];
    const float* bf = (const float*)d_in[3];
    const float* Wi = (const float*)d_in[4];
    const float* bi = (const float*)d_in[5];
    const float* Wh = (const float*)d_in[6];
    const float* bh = (const float*)d_in[7];
    float* out = (float*)d_out;

    const size_t S  = (size_t)M_ * D_;        // one gate pre-activation array
    const size_t SC = (size_t)B_ * NC * D_;   // chunk-summary array
    const size_t SW = (size_t)3 * NDIM * KDIM;

    float* wsf    = (float*)d_ws;
    float* kf     = wsf;
    float* ki     = wsf + S;
    float* kh     = wsf + 2 * S;
    float* Asum   = wsf + 3 * S;
    float* Bsum   = Asum + SC;
    float* prefix = Bsum + SC;
    bf16_t* wsb   = (bf16_t*)(prefix + SC);
    bf16_t* Xhi   = wsb;
    bf16_t* Xlo   = Xhi + S;
    bf16_t* WThi  = Xlo + S;
    bf16_t* WTlo  = WThi + SW;

    // pre-passes: split-convert X; split-convert + transpose W
    minlstm_convert_x<<<(unsigned)(S / (256 * 8)), 256, 0, stream>>>(x, Xhi, Xlo);
    dim3 gW(KDIM / 32, NDIM / 32, 3);
    minlstm_convert_w<<<gW, 256, 0, stream>>>(Wf, Wi, Wh, WThi, WTlo);

    // WMMA GEMM for the three gate pre-activations
    dim3 gGemm(NDIM / BN, M_ / BM, 3);
    minlstm_gate_gemm<<<gGemm, 256, 0, stream>>>(Xhi, Xlo, WThi, WTlo,
                                                 bf, bi, bh, kf);

    // chunked log-space scan
    dim3 gScan(D_ / 256, NC, B_);
    minlstm_scan_chunk<<<gScan, 256, 0, stream>>>(kf, ki, kh, Asum, Bsum);
    minlstm_scan_combine<<<(B_ * D_ + 255) / 256, 256, 0, stream>>>(h0, Asum, Bsum, prefix);
    minlstm_scan_apply<<<gScan, 256, 0, stream>>>(kf, ki, kh, prefix, out);
}
